// KNNImputation_31885837206095
// MI455X (gfx1250) — matrix operand order repaired and checked by reading, not compiled, not run on hardware
//
#include <hip/hip_runtime.h>

typedef __attribute__((ext_vector_type(16))) _Float16 v16h;
typedef __attribute__((ext_vector_type(8)))  float    v8f;

#define T1 17
#define NNODES 2048
#define CCH 8                  // output channels (D_in - 1)
#define NTILES (NNODES / 16)   // 128 donor tiles of 16
#define NPAIR (NTILES / 2)     // 64 tile-pairs per wave loop
#define FRAG_HALFS 512         // one 16x32 f16 fragment: 32 lanes * 16 halfs
#define KNN_K 5
#define BIGF 3.0e38f

// ---------------------------------------------------------------------------
// Branchless insert of candidate (d2v, cf, vv) into a 5-entry list sorted
// ascending by (d2/cn), compared via cross-multiplication (all terms >= 0).
// Invalid candidates encoded as (BIGF, 1, *) lose every comparison.
// ---------------------------------------------------------------------------
__device__ __forceinline__ void insert5(float d2v, float cf, float vv,
                                        float (&D)[KNN_K], float (&C)[KNN_K],
                                        float (&V)[KNN_K])
{
    bool lt[KNN_K];
#pragma unroll
    for (int k = 0; k < KNN_K; ++k)
        lt[k] = d2v * C[k] < D[k] * cf;          // cand ranks before slot k

    float nD[KNN_K], nC[KNN_K], nV[KNN_K];
#pragma unroll
    for (int k = KNN_K - 1; k >= 1; --k) {
        nD[k] = lt[k] ? (lt[k - 1] ? D[k - 1] : d2v) : D[k];
        nC[k] = lt[k] ? (lt[k - 1] ? C[k - 1] : cf ) : C[k];
        nV[k] = lt[k] ? (lt[k - 1] ? V[k - 1] : vv ) : V[k];
    }
    nD[0] = lt[0] ? d2v : D[0];
    nC[0] = lt[0] ? cf  : C[0];
    nV[0] = lt[0] ? vv  : V[0];
#pragma unroll
    for (int k = 0; k < KNN_K; ++k) { D[k] = nD[k]; C[k] = nC[k]; V[k] = nV[k]; }
}

// ---------------------------------------------------------------------------
// Kernel 1: pack per-node feature vectors into WMMA fragment layouts.
//   f (A-side, query):  k in [0,17): x^2 ; [20,37): present ; [40,57): -2x
//   g (B-side, donor):  k in [0,17): present ; [20,37): x^2 ; [40,57): x
//   => dot(f_i, g_j) = sq_i.pf_j + pf_i.sq_j - 2 x_i.x_j = d2[i][j]
// ---------------------------------------------------------------------------
__global__ void knn_pack(const float* __restrict__ x_all,
                         const int* __restrict__ mask,
                         _Float16* __restrict__ Astg,
                         _Float16* __restrict__ Bstg,
                         unsigned* __restrict__ presMask,
                         float* __restrict__ xcolv)
{
    int tid = blockIdx.x * blockDim.x + threadIdx.x;
    if (tid >= CCH * NNODES) return;
    int c = tid / NNODES;
    int n = tid % NNODES;

    unsigned pm = 0;
    float xv[T1];
#pragma unroll
    for (int t = 0; t < T1; ++t) {
        bool pres = (mask[t * NNODES + n] == 0);
        float x = x_all[((size_t)t * NNODES + n) * (CCH + 1) + c];
        pm |= (pres ? 1u : 0u) << t;
        xv[t] = pres ? x : 0.0f;   // Xz: zero at missing entries
    }
    if (c == 0) presMask[n] = pm;
    xcolv[(size_t)c * NNODES + n] = xv[T1 - 1];   // donor value at t=16 (0 if absent)

    auto valF = [&](int k) -> float {
        if (k < T1)                 return xv[k] * xv[k];
        if (k >= 20 && k < 20 + T1) return ((pm >> (k - 20)) & 1u) ? 1.0f : 0.0f;
        if (k >= 40 && k < 40 + T1) return -2.0f * xv[k - 40];
        return 0.0f;
    };
    auto valG = [&](int k) -> float {
        if (k < T1)                 return ((pm >> k) & 1u) ? 1.0f : 0.0f;
        if (k >= 20 && k < 20 + T1) { float x = xv[k - 20]; return x * x; }
        if (k >= 40 && k < 40 + T1) return xv[k - 40];
        return 0.0f;
    };

    int tile = n >> 4;
    int r = n & 15;
#pragma unroll
    for (int h = 0; h < 2; ++h) {
        // A fragment (16x32 f16): lanes 0-15 row=r K{0..7,16..23}; lanes 16-31 K{8..15,24..31}
        _Float16* baseA = Astg + (((size_t)(c * NTILES + tile) * 2 + h) * FRAG_HALFS);
#pragma unroll
        for (int s = 0; s < 16; ++s) {
            int kLo = h * 32 + (s < 8 ? s : 8 + s);
            int kHi = h * 32 + (s < 8 ? 8 + s : 16 + s);
            baseA[r * 16 + s]        = (_Float16)valF(kLo);
            baseA[(r + 16) * 16 + s] = (_Float16)valF(kHi);
        }
        // B fragment (32x16 f16): lane = column; lanes 0-15 K{0..7,16..23}; lanes 16-31 K{8..15,24..31}
        _Float16* baseB = Bstg + (((size_t)(c * NTILES + tile) * 2 + h) * FRAG_HALFS);
#pragma unroll
        for (int s = 0; s < 16; ++s) {
            baseB[r * 16 + s]        = (_Float16)valG(h * 32 + (s < 8 ? s : 8 + s));
            baseB[(r + 16) * 16 + s] = (_Float16)valG(h * 32 + (s < 8 ? 8 + s : 16 + s));
        }
    }
}

// ---------------------------------------------------------------------------
// Kernel 2: observed column mean per channel (sklearn fallback).
// ---------------------------------------------------------------------------
__global__ void knn_colmean(const unsigned* __restrict__ presMask,
                            const float* __restrict__ xcolv,
                            float* __restrict__ colmean)
{
    __shared__ float ssum[256];
    __shared__ float scnt[256];
    int c = blockIdx.x;
    float s = 0.0f, k = 0.0f;
    for (int n = threadIdx.x; n < NNODES; n += blockDim.x) {
        s += xcolv[(size_t)c * NNODES + n];               // already 0 if absent
        k += (float)((presMask[n] >> (T1 - 1)) & 1u);
    }
    ssum[threadIdx.x] = s; scnt[threadIdx.x] = k;
    __syncthreads();
    for (int off = 128; off > 0; off >>= 1) {
        if ((int)threadIdx.x < off) {
            ssum[threadIdx.x] += ssum[threadIdx.x + off];
            scnt[threadIdx.x] += scnt[threadIdx.x + off];
        }
        __syncthreads();
    }
    if (threadIdx.x == 0) colmean[c] = ssum[0] / fmaxf(scnt[0], 1.0f);
}

// ---------------------------------------------------------------------------
// Kernel 3: fused distance (WMMA f16 -> f32) + streaming branchless top-5.
// One wave32 per (channel, 16-query-row tile). Each iteration computes TWO
// donor tiles (4 WMMAs); lanes 0-15 scan tile A's columns, lanes 16-31 scan
// tile B's, each keeping a private top-5 for query row (lane & 15).
// ---------------------------------------------------------------------------
__global__ void __launch_bounds__(32)
knn_main(const _Float16* __restrict__ Astg, const _Float16* __restrict__ Bstg,
         const unsigned* __restrict__ presMask, const float* __restrict__ xcolv,
         const float* __restrict__ colmean, float* __restrict__ out)
{
    int c    = blockIdx.x / NTILES;
    int rt   = blockIdx.x % NTILES;
    int lane = threadIdx.x;
    int r    = lane & 15;        // query row handled by this lane
    int half = lane >> 4;        // 0: even donor tile, 1: odd donor tile

    __shared__ float    distT[16][33];   // [query row][donor col 0..31], +1 pad
    __shared__ unsigned maskT[32];
    __shared__ float    valT[32];
    __shared__ float    mrgD2[16][KNN_K];
    __shared__ float    mrgCn[16][KNN_K];
    __shared__ float    mrgV [16][KNN_K];

    const _Float16* aBase = Astg + (size_t)(c * NTILES + rt) * 2 * FRAG_HALFS;
    v16h a0 = *(const v16h*)(aBase + lane * 16);
    v16h a1 = *(const v16h*)(aBase + FRAG_HALFS + lane * 16);

    unsigned qmask = presMask[rt * 16 + r];
    const float* xcolvC = xcolv + (size_t)c * NNODES;

    float bD2[KNN_K] = {BIGF, BIGF, BIGF, BIGF, BIGF};
    float bCn[KNN_K] = {1.f, 1.f, 1.f, 1.f, 1.f};
    float bV [KNN_K] = {0.f, 0.f, 0.f, 0.f, 0.f};

    const _Float16* bTileBase = Bstg + (size_t)c * NTILES * 2 * FRAG_HALFS;
    const float* drow = &distT[r][0];

    for (int it = 0; it < NPAIR; ++it) {
        const _Float16* bBase = bTileBase + (size_t)it * 4 * FRAG_HALFS;
        v16h b0 = *(const v16h*)(bBase + 0 * FRAG_HALFS + lane * 16);
        v16h b1 = *(const v16h*)(bBase + 1 * FRAG_HALFS + lane * 16);
        v16h b2 = *(const v16h*)(bBase + 2 * FRAG_HALFS + lane * 16);
        v16h b3 = *(const v16h*)(bBase + 3 * FRAG_HALFS + lane * 16);
        if (it + 1 < NPAIR)
            __builtin_prefetch(bBase + 4 * FRAG_HALFS + lane * 16, 0, 3);

        v8f acc0 = {};
        acc0 = __builtin_amdgcn_wmma_f32_16x16x32_f16(false, a0, false, b0,
                                                      (short)0, acc0, false, false);
        acc0 = __builtin_amdgcn_wmma_f32_16x16x32_f16(false, a1, false, b1,
                                                      (short)0, acc0, false, false);
        v8f acc1 = {};
        acc1 = __builtin_amdgcn_wmma_f32_16x16x32_f16(false, a0, false, b2,
                                                      (short)0, acc1, false, false);
        acc1 = __builtin_amdgcn_wmma_f32_16x16x32_f16(false, a1, false, b3,
                                                      (short)0, acc1, false, false);

        // C/D layout: VGPR i, lanes0-15 -> M=i, N=lane; lanes16-31 -> M=i+8, N=lane-16
        int rbase = half * 8;
#pragma unroll
        for (int i = 0; i < 8; ++i) {
            distT[rbase + i][r]      = acc0[i];
            distT[rbase + i][16 + r] = acc1[i];
        }
        {
            int j = it * 32 + lane;        // all 32 donors of this tile pair
            maskT[lane] = presMask[j];
            valT[lane]  = xcolvC[j];
        }
        __syncthreads();

        // full-wave branchless selection: lane scans 16 cols of its half-tile
#pragma unroll 4
        for (int s = 0; s < 16; ++s) {
            int cidx = half * 16 + s;
            unsigned dm = maskT[cidx];
            unsigned iv = qmask & dm;
            bool valid = (((dm >> (T1 - 1)) & 1u) != 0u) & (iv != 0u);
            float cf  = valid ? (float)__popc(iv) : 1.0f;
            float d2v = valid ? fmaxf(drow[cidx], 0.0f) : BIGF;
            insert5(d2v, cf, valT[cidx], bD2, bCn, bV);
        }
        __syncthreads();
    }

    // merge: high half publishes its lists, low half folds them in
    if (lane >= 16) {
#pragma unroll
        for (int t = 0; t < KNN_K; ++t) {
            mrgD2[r][t] = bD2[t]; mrgCn[r][t] = bCn[t]; mrgV[r][t] = bV[t];
        }
    }
    __syncthreads();

    if (lane < 16) {
#pragma unroll
        for (int t = 0; t < KNN_K; ++t)   // stale (BIGF,1,0) slots self-neutralize
            insert5(mrgD2[r][t], mrgCn[r][t], mrgV[r][t], bD2, bCn, bV);

        int i = rt * 16 + r;
        float res;
        if ((qmask >> (T1 - 1)) & 1u) {
            res = xcolvC[i];                              // observed: keep value
        } else {
            float sum = 0.0f, k = 0.0f;
#pragma unroll
            for (int t = 0; t < KNN_K; ++t)
                if (bD2[t] < BIGF) { sum += bV[t]; k += 1.0f; }
            res = (k > 0.0f) ? (sum / k) : colmean[c];    // sklearn fallback
        }
        out[(size_t)i * CCH + c] = res;
    }
}

// ---------------------------------------------------------------------------
extern "C" void kernel_launch(void* const* d_in, const int* in_sizes, int n_in,
                              void* d_out, int out_size, void* d_ws, size_t ws_size,
                              hipStream_t stream)
{
    const float* x_all = (const float*)d_in[0];   // [17, 2048, 9] f32
    const int*   mask  = (const int*)d_in[1];     // [17, 2048] i32
    float* out = (float*)d_out;                   // [2048, 8] f32

    char* ws = (char*)d_ws;
    size_t szFrag = (size_t)CCH * NTILES * 2 * FRAG_HALFS * sizeof(_Float16); // 2 MB
    _Float16* Astg     = (_Float16*)(ws);
    _Float16* Bstg     = (_Float16*)(ws + szFrag);
    unsigned* presMask = (unsigned*)(ws + 2 * szFrag);
    float*    xcolv    = (float*)(ws + 2 * szFrag + NNODES * sizeof(unsigned));
    float*    colmean  = (float*)(ws + 2 * szFrag + NNODES * sizeof(unsigned)
                                  + (size_t)CCH * NNODES * sizeof(float));

    int packThreads = CCH * NNODES;               // 16384
    knn_pack<<<(packThreads + 255) / 256, 256, 0, stream>>>(
        x_all, mask, Astg, Bstg, presMask, xcolv);

    knn_colmean<<<CCH, 256, 0, stream>>>(presMask, xcolv, colmean);

    knn_main<<<CCH * NTILES, 32, 0, stream>>>(
        Astg, Bstg, presMask, xcolv, colmean, out);
}
